// IsoMaxPlusLossFirstPart_4518305595445
// MI455X (gfx1250) — compile-verified, matrix operand order checked by reading, not run on hardware
//
#include <hip/hip_runtime.h>
#include <math.h>

// ---------------- CDNA5 async global->LDS support (guarded) ----------------
#if __has_builtin(__builtin_amdgcn_global_load_async_to_lds_b128)
#define HAVE_ASYNC_LDS 1
#else
#define HAVE_ASYNC_LDS 0
#warning "PROBE: async_to_lds builtin NOT available; using reg-staged copies"
#endif

typedef __attribute__((ext_vector_type(16))) _Float16 v16h;
typedef __attribute__((ext_vector_type(8)))  _Float16 v8h;
typedef __attribute__((ext_vector_type(4)))  _Float16 v4h;
typedef __attribute__((ext_vector_type(8)))  float    v8f;
typedef __attribute__((ext_vector_type(4)))  int      v4i;

#define TILE_M 128
#define TILE_N 128
#define TILE_K 64
// LDS row pitch in halves: 72 halves = 144 B. 16B-aligned (b128 legal) and
// 144 B = 36 banks -> 36*r mod 64 distinct for r = 0..15 (conflict-free).
#define LDT    72

// 16-byte global->LDS copy: async on CDNA5, reg-staged fallback otherwise.
__device__ __forceinline__ void copy16_to_lds(const _Float16* g, _Float16* l) {
#if HAVE_ASYNC_LDS
  __builtin_amdgcn_global_load_async_to_lds_b128(
      (__attribute__((address_space(1))) v4i*)g,
      (__attribute__((address_space(3))) v4i*)l,
      /*offset=*/0, /*cpol=*/0);
#else
  *(v8h*)l = *(const v8h*)g;
#endif
}

__device__ __forceinline__ void wait_async_copies() {
#if HAVE_ASYNC_LDS
#if __has_builtin(__builtin_amdgcn_s_wait_asynccnt)
  __builtin_amdgcn_s_wait_asynccnt(0);
#else
  asm volatile("s_wait_asynccnt 0x0" ::: "memory");
#endif
#endif
  // Fallback path: ds_store completion is tracked by DScnt and the compiler
  // inserts s_wait_dscnt before the barrier in __syncthreads().
}

// ---------------------------------------------------------------------------
// Kernel 1/2: L2-normalize each D=1024 row of an f32 matrix, emit f16.
// One block per row; 256 threads * float4 = 1024 elements.
// ---------------------------------------------------------------------------
__global__ __launch_bounds__(256) void rownorm_f16_kernel(
    const float* __restrict__ src, _Float16* __restrict__ dst, int D) {
  const int row = blockIdx.x;
  const int t   = threadIdx.x;
  const float4 x = ((const float4*)(src + (size_t)row * D))[t];

  float ss = x.x * x.x + x.y * x.y + x.z * x.z + x.w * x.w;
#pragma unroll
  for (int o = 16; o > 0; o >>= 1) ss += __shfl_xor(ss, o, 32);

  __shared__ float wsum[8];
  if ((t & 31) == 0) wsum[t >> 5] = ss;
  __syncthreads();
  float tot = 0.0f;
#pragma unroll
  for (int i = 0; i < 8; ++i) tot += wsum[i];

  const float inv = 1.0f / fmaxf(sqrtf(tot), 1e-12f);
  v4h h;
  h.x = (_Float16)(x.x * inv);
  h.y = (_Float16)(x.y * inv);
  h.z = (_Float16)(x.z * inv);
  h.w = (_Float16)(x.w * inv);
  ((v4h*)(dst + (size_t)row * D))[t] = h;
}

// ---------------------------------------------------------------------------
// Kernel 3: double-buffered WMMA GEMM (dots) + IsoMax+ epilogue.
//   out[n][c] = -|s| * sqrt(max(2 - 2*dot, 1e-12))
// Block: 256 threads = 8 waves; block tile 128x128, K-tile 64.
// Wave grid 4(M) x 2(N); each wave -> 32x64 via 2x4 v_wmma_f32_16x16x32_f16.
// Pipeline: async-prefetch tile kt+1 into LDS buf^1 while computing buf.
// ---------------------------------------------------------------------------
__global__ __launch_bounds__(256) void isomax_gemm_kernel(
    const _Float16* __restrict__ A,   // [N, D] normalized features (f16)
    const _Float16* __restrict__ B,   // [C, D] normalized prototypes (f16)
    const float* __restrict__ dscale, // [1]
    float* __restrict__ out,          // [N, C]
    int N, int C, int D) {
  __shared__ _Float16 sA[2][TILE_M * LDT];  // 2 x 18,432 B
  __shared__ _Float16 sB[2][TILE_N * LDT];  // 2 x 18,432 B  (73.7 KB total)

  const int tid    = threadIdx.x;
  const int wave   = tid >> 5;
  const int lane   = tid & 31;
  const int lane16 = lane & 15;
  const int hi     = lane >> 4;     // 0: K 0..7/16..23, 1: K 8..15/24..31

  const int wm = wave & 3;          // wave row   (4 * 32 = 128)
  const int wn = wave >> 2;         // wave col   (2 * 64 = 128)

  const int blockM = blockIdx.y * TILE_M;
  const int blockN = blockIdx.x * TILE_N;

  v8f acc[2][4];
#pragma unroll
  for (int mt = 0; mt < 2; ++mt)
#pragma unroll
    for (int nt = 0; nt < 4; ++nt) acc[mt][nt] = (v8f){};

  // Staging: 128 rows x 64 halves = 1024 x 16B chunks per matrix;
  // 256 threads x 4 passes of b128, coalesced (L2-resident operands).
  const int chunk     = tid & 7;     // 16B chunk within a row
  const int rowInPass = tid >> 3;    // 0..31

  const _Float16* gA = A + (size_t)blockM * D + chunk * 8;
  const _Float16* gB = B + (size_t)blockN * D + chunk * 8;

  const int kTiles = D / TILE_K;

  // Prologue: stage tile 0 into buffer 0.
  {
#pragma unroll
    for (int pass = 0; pass < 4; ++pass) {
      const int r = pass * 32 + rowInPass;
      copy16_to_lds(gA + (size_t)r * D, &sA[0][r * LDT + chunk * 8]);
      copy16_to_lds(gB + (size_t)r * D, &sB[0][r * LDT + chunk * 8]);
    }
    wait_async_copies();
    __syncthreads();
  }

  for (int kt = 0; kt < kTiles; ++kt) {
    const int cur = kt & 1;

    // Prefetch tile kt+1 into the other buffer (async: lands in LDS with no
    // VGPR round-trip; tracked by ASYNCcnt).
    if (kt + 1 < kTiles) {
      const int kcol = (kt + 1) * TILE_K;
#pragma unroll
      for (int pass = 0; pass < 4; ++pass) {
        const int r = pass * 32 + rowInPass;
        copy16_to_lds(gA + (size_t)r * D + kcol,
                      &sA[cur ^ 1][r * LDT + chunk * 8]);
        copy16_to_lds(gB + (size_t)r * D + kcol,
                      &sB[cur ^ 1][r * LDT + chunk * 8]);
      }
    }

    // Compute on the current buffer.
    const _Float16* tA = sA[cur];
    const _Float16* tB = sB[cur];
#pragma unroll
    for (int s = 0; s < 2; ++s) {  // two K=32 WMMA steps per K-tile
      const int cb = s * 32 + hi * 8;
      v16h af[2], bf[4];
#pragma unroll
      for (int mt = 0; mt < 2; ++mt) {
        const _Float16* p = tA + (wm * 32 + mt * 16 + lane16) * LDT + cb;
        union { v16h v; v8h h[2]; } u;
        u.h[0] = *(const v8h*)(p);        // K = cb .. cb+7
        u.h[1] = *(const v8h*)(p + 16);   // K = cb+16 .. cb+23
        af[mt] = u.v;
      }
#pragma unroll
      for (int nt = 0; nt < 4; ++nt) {
        const _Float16* p = tB + (wn * 64 + nt * 16 + lane16) * LDT + cb;
        union { v16h v; v8h h[2]; } u;
        u.h[0] = *(const v8h*)(p);
        u.h[1] = *(const v8h*)(p + 16);
        bf[nt] = u.v;
      }
#pragma unroll
      for (int mt = 0; mt < 2; ++mt)
#pragma unroll
        for (int nt = 0; nt < 4; ++nt)
          acc[mt][nt] = __builtin_amdgcn_wmma_f32_16x16x32_f16(
              false, af[mt], false, bf[nt], (short)0, acc[mt][nt],
              false, false);
    }

    // Prefetch must be complete (per wave), then all waves rendezvous before
    // anyone reads the freshly filled buffer / overwrites the current one.
    wait_async_copies();
    __syncthreads();
  }

  // Epilogue + non-temporal stores (keep the 512 MB output out of L2,
  // where the f16 operands are resident).
  const float dsc = fabsf(dscale[0]);
#pragma unroll
  for (int mt = 0; mt < 2; ++mt) {
#pragma unroll
    for (int nt = 0; nt < 4; ++nt) {
      const int col = blockN + wn * 64 + nt * 16 + lane16;
#pragma unroll
      for (int v = 0; v < 8; ++v) {
        const int row = blockM + wm * 32 + mt * 16 + hi * 8 + v;
        const float dot   = acc[mt][nt][v];
        const float sq    = fmaxf(2.0f - 2.0f * dot, 1e-12f);
        const float logit = -dsc * __builtin_amdgcn_sqrtf(sq);  // v_sqrt_f32
        __builtin_nontemporal_store(logit, out + (size_t)row * C + col);
      }
    }
  }
}

// ---------------------------------------------------------------------------
extern "C" void kernel_launch(void* const* d_in, const int* in_sizes, int n_in,
                              void* d_out, int out_size, void* d_ws,
                              size_t ws_size, hipStream_t stream) {
  const float* features   = (const float*)d_in[0];
  const float* prototypes = (const float*)d_in[1];
  const float* dscale     = (const float*)d_in[2];
  float*       out        = (float*)d_out;

  const int D = 1024;
  const int N = in_sizes[0] / D;   // 16384
  const int C = in_sizes[1] / D;   // 8192

  // Workspace layout: [0, N*D) f16 normalized features,
  //                   [N*D, N*D + C*D) f16 normalized prototypes. (48 MB)
  _Float16* Af16 = (_Float16*)d_ws;
  _Float16* Bf16 = Af16 + (size_t)N * D;

  rownorm_f16_kernel<<<N, 256, 0, stream>>>(features, Af16, D);
  rownorm_f16_kernel<<<C, 256, 0, stream>>>(prototypes, Bf16, D);

  dim3 grid(C / TILE_N, N / TILE_M);  // (64, 128)
  isomax_gemm_kernel<<<grid, 256, 0, stream>>>(Af16, Bf16, dscale, out,
                                               N, C, D);
}
